// HME_54047868453085
// MI455X (gfx1250) — matrix operand order repaired and checked by reading, not compiled, not run on hardware
//
#include <hip/hip_runtime.h>
#include <math.h>

typedef __bf16 bf16_t;
typedef __attribute__((ext_vector_type(2)))  __bf16 v2bf;
typedef __attribute__((ext_vector_type(8)))  __bf16 v8bf;
typedef __attribute__((ext_vector_type(16))) __bf16 v16bf;
typedef __attribute__((ext_vector_type(8)))  float  v8f;

#define WMMA_BF16(a, b, c) \
    __builtin_amdgcn_wmma_f32_16x16x32_bf16(false, (a), false, (b), (short)0, (c), false, false)
#define AFRAG(lo, hi) __builtin_shufflevector((lo), (hi), 0,1,2,3,4,5,6,7,8,9,10,11,12,13,14,15)

constexpr int C_SENT = 690, C_ENT = 230, C_HIE = 512;
constexpr int KTOT = 920, KPAD = 928, KSTR = 960;   // KSTR: +1 K-tile pad for B preload
constexpr int XSTR = 936, ZSTR = 520, NESTR = 72;   // LDS strides
constexpr int N_L0 = 53, N_L1 = 36, N_L2 = 6, N_REL = 95;
constexpr int MB = 64;                              // rows per block

// ---- workspace layout (bytes) ----
constexpr size_t WS_WT   = 0;                                   // bf16 [512][960]  W_fus^T, K-padded
constexpr size_t WS_GT   = WS_WT   + (size_t)C_HIE * KSTR * 2;  // bf16 [16][960]   gate bag weights
constexpr size_t WS_RNB  = WS_GT   + (size_t)16 * KSTR * 2;     // bf16 [96][512]   rel_embs (row 95 = 0)
constexpr size_t WS_RINV = WS_RNB  + (size_t)96 * C_HIE * 2;    // f32  [96]
constexpr size_t WS_CDOT = WS_RINV + (size_t)96 * 4;            // f32  [96][6]

// ---- dynamic LDS layout (bytes), lifetime-overlapped ----
constexpr int SM_Z  = 0;                            // bf16 [3][64][520]    (phase 3..4)
constexpr int SM_X  = 0;                            // bf16 [64][936]       (phase 0..1, inside Z)
constexpr int SM_NE = 3 * MB * ZSTR * 2;            // bf16 [512][72] NE^T  (phase 1..3)  = 199680
constexpr int SM_P  = SM_NE;                        // f32  [64][96]        (phase 4..5, inside NE)
constexpr int SM_BD = SM_NE + C_HIE * NESTR * 2;    // f32 [64][16] / int par stage = 273408
constexpr int SM_OG = SM_BD + MB * 16 * 4;          // f32  [3][64]         = 277504
constexpr int SM_ZN = SM_OG + 3 * MB * 4;           // f32  [3][64]         = 278272
constexpr int SM_TOTAL = SM_ZN + 3 * MB * 4;        // = 279040  (1 block/WGP)

__device__ __forceinline__ float sigm(float x) { return 1.f / (1.f + expf(-x)); }

// ============ prep kernels ============

__global__ void k_prep_wt(const float* __restrict__ Wfus, bf16_t* __restrict__ Wt) {
    int i = blockIdx.x * 256 + threadIdx.x;       // over 512*960
    int n = i / KSTR, k = i - n * KSTR;
    Wt[(size_t)n * KSTR + k] = (k < KTOT) ? (bf16_t)Wfus[(size_t)k * C_HIE + n] : (bf16_t)0.f;
}

__global__ void k_prep_gt(const float* Wig1, const float* Wog1,
                          const float* Wig2, const float* Wog2,
                          const float* Wig3, const float* Wog3,
                          bf16_t* __restrict__ Gt) {
    int i = blockIdx.x * 256 + threadIdx.x;       // over 16*960
    int g = i / KSTR, k = i - g * KSTR;
    const float* sel[6] = {Wig1, Wog1, Wig2, Wog2, Wig3, Wog3};
    float v = 0.f;
    if (g < 6 && k < C_SENT) v = sel[g][k];
    Gt[(size_t)g * KSTR + k] = (bf16_t)v;
}

__global__ void k_prep_cell(const float* __restrict__ cell,
                            const float* Wig1, const float* Wog1,
                            const float* Wig2, const float* Wog2,
                            const float* Wig3, const float* Wog3,
                            float* __restrict__ cdot) {
    int r = blockIdx.x / 6, g = blockIdx.x % 6;   // 96*6 blocks of 64
    const float* sel[6] = {Wig1, Wog1, Wig2, Wog2, Wig3, Wog3};
    const float* w = sel[g] + C_SENT;
    __shared__ float red[64];
    float s = 0.f;
    for (int k = threadIdx.x; k < C_HIE; k += 64) s += cell[(size_t)r * C_HIE + k] * w[k];
    red[threadIdx.x] = s;
    __syncthreads();
    for (int o = 32; o; o >>= 1) { if (threadIdx.x < o) red[threadIdx.x] += red[threadIdx.x + o]; __syncthreads(); }
    if (threadIdx.x == 0) cdot[r * 6 + g] = red[0];
}

__global__ void k_prep_rel(const float* __restrict__ rel, bf16_t* __restrict__ rnb,
                           float* __restrict__ rinv) {
    int n = blockIdx.x;   // 0..95
    __shared__ float red[64];
    float s = 0.f;
    for (int k = threadIdx.x; k < C_HIE; k += 64) {
        float v = (n < N_REL) ? rel[(size_t)n * C_HIE + k] : 0.f;
        rnb[(size_t)n * C_HIE + k] = (bf16_t)v;
        s += v * v;
    }
    red[threadIdx.x] = s;
    __syncthreads();
    for (int o = 32; o; o >>= 1) { if (threadIdx.x < o) red[threadIdx.x] += red[threadIdx.x + o]; __syncthreads(); }
    if (threadIdx.x == 0) rinv[n] = (n < N_REL) ? 1.f / fmaxf(sqrtf(red[0]), 1e-8f) : 0.f;
}

// ============ fused main kernel: 64 rows / block, 16 waves ============

__global__ __launch_bounds__(512)
void hme_main(const float* __restrict__ bag, const float* __restrict__ head,
              const float* __restrict__ tail, const float* __restrict__ bfus,
              const float* __restrict__ Wog1, const float* __restrict__ Wog2,
              const float* __restrict__ Wog3,
              const float* big1, const float* bog1,
              const float* big2, const float* bog2,
              const float* big3, const float* bog3,
              const float* __restrict__ cell,
              const int* __restrict__ rel1_idx, const int* __restrict__ rel2_idx,
              const int* __restrict__ par0, const int* __restrict__ par1,
              const bf16_t* __restrict__ Wt, const bf16_t* __restrict__ Gt,
              const bf16_t* __restrict__ rnb, const float* __restrict__ rinv,
              const float* __restrict__ cdot, float* __restrict__ out) {
    extern __shared__ char smem[];
    bf16_t* sX  = (bf16_t*)(smem + SM_X);
    bf16_t* sNE = (bf16_t*)(smem + SM_NE);      // transposed: [col 0..511][row 0..63], stride 72
    bf16_t* sZ  = (bf16_t*)(smem + SM_Z);
    float*  sP  = (float*)(smem + SM_P);
    float*  sBD = (float*)(smem + SM_BD);
    float*  sOG = (float*)(smem + SM_OG);
    float*  sZN = (float*)(smem + SM_ZN);

    const int tid  = threadIdx.x;
    const int lane = tid & 31;
    const int w    = __builtin_amdgcn_readfirstlane(tid >> 5);   // scalar wave id 0..15
    const int m0   = blockIdx.x * MB;
    const int arow = lane & 15;
    const int asel = (lane >> 4) ? 16 : 0;     // A frag K sub-offset
    const int bsel = (lane >> 4) ? 16 : 0;     // B frag K sub-offset
    const int rbase = (lane >> 4) ? 8 : 0;     // C frag row base

    // ---- Phase 0: stage X = [bag | head-tail | 0] as bf16 (float2 loads) ----
    {
        const int PR = KPAD / 2;               // 464 pairs per row
        for (int e = tid; e < MB * PR; e += 512) {
            int r = e / PR, c2 = e - r * PR;
            int g = m0 + r;
            float2 v;
            if (c2 < 345) {
                v = *(const float2*)(bag + (size_t)g * C_SENT + 2 * c2);
            } else if (c2 < 460) {
                int cc = 2 * (c2 - 345);
                float2 h = *(const float2*)(head + (size_t)g * C_ENT + cc);
                float2 t = *(const float2*)(tail + (size_t)g * C_ENT + cc);
                v = make_float2(h.x - t.x, h.y - t.y);
            } else {
                v = make_float2(0.f, 0.f);
            }
            v2bf o; o[0] = (bf16_t)v.x; o[1] = (bf16_t)v.y;
            *(v2bf*)(sX + r * XSTR + 2 * c2) = o;
        }
    }
    __syncthreads();

    // ---- Phase 1: new_emb = X @ W_fus ; wave owns 2 N-tiles x 4 M-tiles ----
    v8f acc[8] = {};
    const int n0 = w * 32;
    const bf16_t* wbase = Wt + (size_t)(n0 + arow) * KSTR + bsel;
    v16bf b0 = *(const v16bf*)(wbase);
    v16bf b1 = *(const v16bf*)(wbase + 16 * KSTR);
    for (int kk = 0; kk < KPAD; kk += 32) {
        // preload next K-step B fragments (KSTR padding keeps this in-bounds)
        v16bf p0 = *(const v16bf*)(wbase + kk + 32);
        v16bf p1 = *(const v16bf*)(wbase + 16 * KSTR + kk + 32);
        // load all 4 A fragments first (clause-able), then 8 back-to-back WMMAs
        v16bf a[4];
#pragma unroll
        for (int mt = 0; mt < 4; ++mt) {
            const bf16_t* xp = sX + (mt * 16 + arow) * XSTR + kk + asel;
            a[mt] = AFRAG(*(const v8bf*)xp, *(const v8bf*)(xp + 8));
        }
#pragma unroll
        for (int mt = 0; mt < 4; ++mt) {
            acc[mt * 2 + 0] = WMMA_BF16(a[mt], b0, acc[mt * 2 + 0]);
            acc[mt * 2 + 1] = WMMA_BF16(a[mt], b1, acc[mt * 2 + 1]);
        }
        b0 = p0; b1 = p1;
    }
    // store new_emb (+bias) transposed, one packed b128 per fragment column
#pragma unroll
    for (int nt = 0; nt < 2; ++nt) {
        int n = n0 + nt * 16 + arow;
        float bias = bfus[n];
#pragma unroll
        for (int mt = 0; mt < 4; ++mt) {
            v8f av = acc[mt * 2 + nt];
            v8bf pk;
#pragma unroll
            for (int j = 0; j < 8; ++j) pk[j] = (bf16_t)(av[j] + bias);
            *(v8bf*)(sNE + n * NESTR + mt * 16 + rbase) = pk;
        }
    }
    // gate tile (bag-side gate dots): waves 0..3, M-tile = w, B from Gt
    if (w < 4) {
        v8f ag = {};
        const bf16_t* gbase = Gt + (size_t)arow * KSTR + bsel;
        for (int kk = 0; kk < KPAD; kk += 32) {
            const bf16_t* xp = sX + (w * 16 + arow) * XSTR + kk + asel;
            v16bf a = AFRAG(*(const v8bf*)xp, *(const v8bf*)(xp + 8));
            v16bf bg = *(const v16bf*)(gbase + kk);
            ag = WMMA_BF16(a, bg, ag);
        }
        if (arow < 6)
            for (int j = 0; j < 8; ++j) sBD[(w * 16 + rbase + j) * 16 + arow] = ag[j];
    }
    __syncthreads();

    // ---- Phase 2: per-row og-cell dots + gate scalars (wave -> 4 rows) ----
    for (int ri = 0; ri < 4; ++ri) {
        int r = w * 4 + ri;
        float e1 = 0.f, e2 = 0.f, e3 = 0.f;
        for (int c = lane; c < C_HIE; c += 32) {
            float ne = (float)sNE[c * NESTR + r];
            e1 += ne * Wog1[C_SENT + c];
            e2 += ne * Wog2[C_SENT + c];
            e3 += ne * Wog3[C_SENT + c];
        }
        for (int o = 16; o; o >>= 1) {
            e1 += __shfl_xor(e1, o);
            e2 += __shfl_xor(e2, o);
            e3 += __shfl_xor(e3, o);
        }
        if (lane == 0) {
            int g = m0 + r;
            int cr1 = N_REL, cr2 = rel2_idx[g], cr3 = rel1_idx[g];
            float ig, og;
            ig = sigm(sBD[r * 16 + 0] + cdot[cr1 * 6 + 0] + big1[0]);
            og = sigm(sBD[r * 16 + 1] + ig * e1 + (1.f - ig) * cdot[cr1 * 6 + 1] + bog1[0]);
            sOG[0 * MB + r] = og;
            ig = sigm(sBD[r * 16 + 2] + cdot[cr2 * 6 + 2] + big2[0]);
            og = sigm(sBD[r * 16 + 3] + ig * e2 + (1.f - ig) * cdot[cr2 * 6 + 3] + bog2[0]);
            sOG[1 * MB + r] = og;
            ig = sigm(sBD[r * 16 + 4] + cdot[cr3 * 6 + 4] + big3[0]);
            og = sigm(sBD[r * 16 + 5] + ig * e3 + (1.f - ig) * cdot[cr3 * 6 + 5] + bog3[0]);
            sOG[2 * MB + r] = og;
        }
    }
    if (tid < 3 * MB) sZN[tid] = 0.f;
    __syncthreads();

    // ---- Phase 3: build z1,z2,z3 (bf16, col pairs) + ||z||^2 ----
    {
        int r = tid >> 3, cg = tid & 7;
        int g = m0 + r;
        int crow[3] = {N_REL, rel2_idx[g], rel1_idx[g]};
        float ogv[3] = {sOG[r], sOG[MB + r], sOG[2 * MB + r]};
        float s[3] = {0.f, 0.f, 0.f};
        for (int i = 0; i < 32; ++i) {
            int c = 2 * (cg + 8 * i);          // interleaved pair index (bank-friendly)
            float ne0 = (float)sNE[c * NESTR + r];
            float ne1 = (float)sNE[(c + 1) * NESTR + r];
#pragma unroll
            for (int L = 0; L < 3; ++L) {
                float2 cr = *(const float2*)(cell + (size_t)crow[L] * C_HIE + c);
                float og = ogv[L];
                float z0 = og * cr.x + (1.f - og) * ne0;
                float z1 = og * cr.y + (1.f - og) * ne1;
                v2bf o; o[0] = (bf16_t)z0; o[1] = (bf16_t)z1;
                *(v2bf*)(sZ + (L * MB + r) * ZSTR + c) = o;
                s[L] += z0 * z0 + z1 * z1;
            }
        }
        atomicAdd(&sZN[r], s[0]);
        atomicAdd(&sZN[MB + r], s[1]);
        atomicAdd(&sZN[2 * MB + r], s[2]);
    }
    __syncthreads();

    // ---- Phase 4: cosine GEMM; 32 tile-jobs, 2 interleaved chains per wave ----
    // stage parent tables to LDS (BD region is dead now)
    if (tid < N_L1) ((int*)sBD)[tid] = par1[tid];
    if (tid < N_L0) ((int*)sBD)[64 + tid] = par0[tid];
    {
        const int tabL[8]  = {2, 2, 2, 2, 1, 1, 1, 0};    // z3 x4, z2 x3, z1
        const int tabN[8]  = {0, 16, 32, 48, 48, 64, 80, 80};
        const int tabLo[8] = {0, 0, 0, 0, 53, 53, 53, 89};
        const int tabHi[8] = {53, 53, 53, 53, 89, 89, 89, 95};
        const int mt = (w * 2) >> 3;
        const int j0 = (w * 2) & 7, j1 = (w * 2 + 1) & 7;
        const int L0v = tabL[j0], L1v = tabL[j1];
        const bf16_t* zb0 = sZ + (size_t)(L0v * MB + mt * 16 + arow) * ZSTR + asel;
        const bf16_t* zb1 = sZ + (size_t)(L1v * MB + mt * 16 + arow) * ZSTR + asel;
        const bf16_t* rb0 = rnb + (size_t)(tabN[j0] + arow) * C_HIE + bsel;
        const bf16_t* rb1 = rnb + (size_t)(tabN[j1] + arow) * C_HIE + bsel;
        v8f c0 = {}, c1 = {};
        for (int kk = 0; kk < C_HIE; kk += 32) {
            v16bf a0 = AFRAG(*(const v8bf*)(zb0 + kk), *(const v8bf*)(zb0 + kk + 8));
            v16bf a1 = AFRAG(*(const v8bf*)(zb1 + kk), *(const v8bf*)(zb1 + kk + 8));
            v16bf bb0 = *(const v16bf*)(rb0 + kk);
            v16bf bb1 = *(const v16bf*)(rb1 + kk);
            c0 = WMMA_BF16(a0, bb0, c0);
            c1 = WMMA_BF16(a1, bb1, c1);
        }
        v8f cres[2] = {c0, c1};
        const int jid[2] = {j0, j1};
#pragma unroll
        for (int q = 0; q < 2; ++q) {
            int jq = jid[q];
            int n = tabN[jq] + arow;
            if (n >= tabLo[jq] && n < tabHi[jq]) {
                float rn = rinv[n];
                int L = tabL[jq];
                for (int j = 0; j < 8; ++j) {
                    int r = mt * 16 + rbase + j;
                    float zi = 1.f / fmaxf(sqrtf(sZN[L * MB + r]), 1e-8f);
                    sP[r * 96 + n] = cres[q][j] * zi * rn;
                }
            }
        }
    }
    __syncthreads();

    // ---- Phase 5: tree path scores + softmaxes (one thread per row) ----
    if (tid < MB) {
        int r = tid;
        const int* sPar = (const int*)sBD;
        float* p = sP + r * 96;
        for (int j = 0; j < N_L1; ++j) p[N_L0 + j] += p[sPar[j]];       // parent in [89,95)
        for (int j = 0; j < N_L0; ++j) p[j]        += p[sPar[64 + j]];  // parent in [53,89)
        float* o = out + (size_t)(m0 + r) * N_REL;
        const int base[3] = {0, N_L0, N_L0 + N_L1};
        const int cnt[3]  = {N_L0, N_L1, N_L2};
        const int poff[3] = {0, N_L0, 89};
        for (int s = 0; s < 3; ++s) {
            const float* q = p + poff[s];
            float m = q[0];
            for (int j = 1; j < cnt[s]; ++j) m = fmaxf(m, q[j]);
            float sum = 0.f;
            for (int j = 0; j < cnt[s]; ++j) sum += expf(q[j] - m);
            float inv = 1.f / sum;
            for (int j = 0; j < cnt[s]; ++j) o[base[s] + j] = expf(q[j] - m) * inv;
        }
    }
}

// ============ launcher ============

extern "C" void kernel_launch(void* const* d_in, const int* in_sizes, int n_in,
                              void* d_out, int out_size, void* d_ws, size_t ws_size,
                              hipStream_t stream) {
    const float* bag  = (const float*)d_in[0];
    const float* head = (const float*)d_in[1];
    const float* tail = (const float*)d_in[2];
    const float* Wfus = (const float*)d_in[3];
    const float* bfus = (const float*)d_in[4];
    const float* Wig1 = (const float*)d_in[5];
    const float* big1 = (const float*)d_in[6];
    const float* Wog1 = (const float*)d_in[7];
    const float* bog1 = (const float*)d_in[8];
    const float* Wig2 = (const float*)d_in[9];
    const float* big2 = (const float*)d_in[10];
    const float* Wog2 = (const float*)d_in[11];
    const float* bog2 = (const float*)d_in[12];
    const float* Wig3 = (const float*)d_in[13];
    const float* big3 = (const float*)d_in[14];
    const float* Wog3 = (const float*)d_in[15];
    const float* bog3 = (const float*)d_in[16];
    const float* cell = (const float*)d_in[17];
    const float* rele = (const float*)d_in[18];
    const int* rel1   = (const int*)d_in[19];
    const int* rel2   = (const int*)d_in[20];
    const int* par0   = (const int*)d_in[21];
    const int* par1   = (const int*)d_in[22];

    char* ws = (char*)d_ws;
    bf16_t* Wt   = (bf16_t*)(ws + WS_WT);
    bf16_t* Gt   = (bf16_t*)(ws + WS_GT);
    bf16_t* rnb  = (bf16_t*)(ws + WS_RNB);
    float*  rinv = (float*)(ws + WS_RINV);
    float*  cdot = (float*)(ws + WS_CDOT);

    k_prep_wt<<<(C_HIE * KSTR) / 256, 256, 0, stream>>>(Wfus, Wt);
    k_prep_gt<<<(16 * KSTR) / 256, 256, 0, stream>>>(Wig1, Wog1, Wig2, Wog2, Wig3, Wog3, Gt);
    k_prep_cell<<<96 * 6, 64, 0, stream>>>(cell, Wig1, Wog1, Wig2, Wog2, Wig3, Wog3, cdot);
    k_prep_rel<<<96, 64, 0, stream>>>(rele, rnb, rinv);

    int B = in_sizes[0] / C_SENT;
    hme_main<<<B / MB, 512, SM_TOTAL, stream>>>(
        bag, head, tail, bfus, Wog1, Wog2, Wog3,
        big1, bog1, big2, bog2, big3, bog3,
        cell, rel1, rel2, par0, par1,
        Wt, Gt, rnb, rinv, cdot, (float*)d_out);
}
